// Actor_50740743635046
// MI455X (gfx1250) — compile-verified, matrix operand order checked by reading, not run on hardware
//
#include <hip/hip_runtime.h>
#include <hip/hip_bf16.h>
#include <math.h>

// ---------------------------------------------------------------- types
typedef __attribute__((ext_vector_type(16))) __bf16       v16bf;
typedef __attribute__((ext_vector_type(8)))  float        v8f;
typedef __attribute__((ext_vector_type(8)))  unsigned int v8u;

static __device__ __forceinline__ unsigned short f2bf(float f) {
  unsigned int u = __builtin_bit_cast(unsigned int, f);
  u += 0x7FFFu + ((u >> 16) & 1u);           // round-to-nearest-even
  return (unsigned short)(u >> 16);
}
static __device__ __forceinline__ float bf2f(unsigned short h) {
  unsigned int u = ((unsigned int)h) << 16;
  return __builtin_bit_cast(float, u);
}

// CDNA5 async copy: ASYNCcnt-tracked DMA global->LDS, 16B per lane.
// ISA pseudocode adds INST_OFFSET to BOTH the LDS and the global address,
// so chunk offsets can live in the offset: immediate.
#define ASYNC_B128(lds32, gaddr, off)                                        \
  asm volatile("global_load_async_to_lds_b128 %0, %1, off offset:%c2"        \
               :: "v"(lds32), "v"(gaddr), "i"(off) : "memory")
#define WAIT_ASYNC(n) asm volatile("s_wait_asynccnt %c0" :: "i"(n) : "memory")

// low 32 bits of a generic pointer to LDS == wave-relative LDS byte address
static __device__ __forceinline__ unsigned ldsAddr(const void* p) {
  return (unsigned)(size_t)p;
}

// ---------------------------------------------------------------- sizes
#define BATCH   65536
#define NNODE   49
#define NEDGE   392
#define K1      588      // 12*49
#define K1P     640      // padded to multiple of 64 (BK)
#define MLPD    512
#define ADIM    3

// workspace layout (byte offsets, 256-aligned)
#define OFF_CSROFF   ((size_t)0)
#define OFF_CSRLIST  ((size_t)256)
#define OFF_CNTINV   ((size_t)2048)
#define OFF_WT1      ((size_t)4096)                          // 512*640*2
#define OFF_WT2      (OFF_WT1 + (size_t)MLPD * K1P * 2)      // 512*512*2
#define OFF_H        (OFF_WT2 + (size_t)MLPD * MLPD * 2)     // B*640*2
#define OFF_C1       (OFF_H   + (size_t)BATCH * K1P * 2)     // B*512*2
#define OFF_C2       OFF_H                                   // H dead after fc1

// ---------------------------------------------------------------- kernel 0: CSR of the (shared) graph
__global__ void build_csr(const long long* __restrict__ ei,
                          int* __restrict__ off, int* __restrict__ lst,
                          float* __restrict__ cntinv) {
  if (blockIdx.x != 0 || threadIdx.x != 0) return;
  int cnt[NNODE];
  for (int i = 0; i < NNODE; ++i) cnt[i] = 0;
  for (int j = 0; j < NEDGE; ++j) cnt[(int)ei[NEDGE + j]]++;   // dst row
  off[0] = 0;
  for (int i = 0; i < NNODE; ++i) off[i + 1] = off[i] + cnt[i];
  int pos[NNODE];
  for (int i = 0; i < NNODE; ++i) pos[i] = off[i];
  for (int j = 0; j < NEDGE; ++j) {
    int d = (int)ei[NEDGE + j];
    lst[pos[d]++] = (int)ei[j];                                 // src row
  }
  for (int i = 0; i < NNODE; ++i)
    cntinv[i] = 1.0f / fmaxf((float)cnt[i], 1.0f);
}

// ---------------------------------------------------------------- kernel 1: weights -> bf16, transposed (Bt[n][k]), K padded
__global__ __launch_bounds__(256) void convert_weights(
    const float* __restrict__ fc1_w,   // (588,512)
    const float* __restrict__ fc2_w,   // (512,512)
    unsigned short* __restrict__ Wt1,  // (512,640)
    unsigned short* __restrict__ Wt2)  // (512,512)
{
  int idx = blockIdx.x * 256 + threadIdx.x;
  const int T1 = MLPD * K1P;
  if (idx < T1) {
    int n = idx / K1P, k = idx % K1P;
    float v = (k < K1) ? fc1_w[(size_t)k * MLPD + n] : 0.0f;
    Wt1[idx] = f2bf(v);
  } else if (idx < T1 + MLPD * MLPD) {
    int i2 = idx - T1;
    int n = i2 / MLPD, k = i2 % MLPD;
    Wt2[i2] = f2bf(fc2_w[(size_t)k * MLPD + n]);
  }
}

// ---------------------------------------------------------------- kernel 2: 3x SAGE conv -> H (bf16, B x 640)
__global__ __launch_bounds__(64) void gnn_kernel(
    const float* __restrict__ x,
    const int* __restrict__ coff, const int* __restrict__ clst,
    const float* __restrict__ cinv,
    const float* __restrict__ W1l, const float* __restrict__ W1r, const float* __restrict__ b1,
    const float* __restrict__ W2l, const float* __restrict__ W2r, const float* __restrict__ b2,
    const float* __restrict__ W3l, const float* __restrict__ W3r, const float* __restrict__ b3,
    unsigned short* __restrict__ H)
{
  __shared__ float sx[NNODE];
  __shared__ float sh1[NNODE][6];
  __shared__ float sh2[NNODE][6];
  const int b = blockIdx.x;
  const int t = threadIdx.x;

  if (t < NNODE) sx[t] = x[(size_t)b * NNODE + t];
  __syncthreads();

  int beg = 0, end = 0;
  float ci = 0.0f;
  if (t < NNODE) { beg = coff[t]; end = coff[t + 1]; ci = cinv[t]; }

  // layer 1: 1 -> 6
  if (t < NNODE) {
    float s = 0.0f;
    for (int e = beg; e < end; ++e) s += sx[clst[e]];
    float mean = s * ci, xv = sx[t];
    #pragma unroll
    for (int f = 0; f < 6; ++f)
      sh1[t][f] = fmaxf(mean * W1l[f] + xv * W1r[f] + b1[f], 0.0f);
  }
  __syncthreads();

  // layer 2: 6 -> 6
  float h2r[6];
  if (t < NNODE) {
    float m[6] = {0,0,0,0,0,0};
    for (int e = beg; e < end; ++e) {
      int s = clst[e];
      #pragma unroll
      for (int f = 0; f < 6; ++f) m[f] += sh1[s][f];
    }
    #pragma unroll
    for (int f = 0; f < 6; ++f) m[f] *= ci;
    #pragma unroll
    for (int f2 = 0; f2 < 6; ++f2) {
      float a = b2[f2];
      #pragma unroll
      for (int f = 0; f < 6; ++f)
        a += m[f] * W2l[f * 6 + f2] + sh1[t][f] * W2r[f * 6 + f2];
      h2r[f2] = fmaxf(a, 0.0f);
    }
  }
  __syncthreads();
  if (t < NNODE) {
    #pragma unroll
    for (int f = 0; f < 6; ++f) sh2[t][f] = h2r[f];
  }
  __syncthreads();

  // layer 3: 6 -> 12 (no relu), write bf16 to H[b][t*12+f3]
  if (t < NNODE) {
    float m[6] = {0,0,0,0,0,0};
    for (int e = beg; e < end; ++e) {
      int s = clst[e];
      #pragma unroll
      for (int f = 0; f < 6; ++f) m[f] += sh2[s][f];
    }
    #pragma unroll
    for (int f = 0; f < 6; ++f) m[f] *= ci;
    unsigned short* hrow = H + (size_t)b * K1P + t * 12;
    #pragma unroll
    for (int f3 = 0; f3 < 12; ++f3) {
      float a = b3[f3];
      #pragma unroll
      for (int f = 0; f < 6; ++f)
        a += m[f] * W3l[f * 12 + f3] + sh2[t][f] * W3r[f * 12 + f3];
      hrow[f3] = f2bf(a);
    }
  }
  // zero the K padding (cols 588..639)
  if (t < (K1P - K1)) H[(size_t)b * K1P + K1 + t] = 0;
}

// ---------------------------------------------------------------- kernel 3: bf16 WMMA GEMM, async-pipelined
// A: M x lda (bf16 row-major), Bt: N x ldb (bf16, Bt[n][k] = B[k][n])
#define BM 128
#define BN 128
#define BK 64
#define ROWDW 36                    // dwords per LDS row: 32 data + 4 pad (16B-aligned, conflict-free)
#define TILEDW (BM * ROWDW)         // one tile (A or B)
#define STAGEDW (2 * TILEDW)        // A+B per pipeline stage
#define NSTG 3                      // pipeline depth-2 lookahead

__global__ __launch_bounds__(256) void gemm_bf16(
    const unsigned short* __restrict__ A,
    const unsigned short* __restrict__ Bt,
    const float* __restrict__ bias,
    unsigned short* __restrict__ C,
    int K, int lda, int ldb, int ldc, int relu)
{
  __shared__ unsigned int sh[NSTG * STAGEDW];   // 3 x 36KB = 108KB (<320KB/WGP)

  const int bm = blockIdx.y * BM;
  const int bn = blockIdx.x * BN;
  const int t    = threadIdx.x;
  const int lane = t & 31;
  const int w    = t >> 5;          // wave 0..7
  const int wm   = (w & 3) * 32;    // wave row offset in tile
  const int wn   = (w >> 2) * 64;   // wave col offset in tile
  const int hl   = lane >> 4;       // K-half select (ISA 16-bit A layout)
  const int ml   = lane & 15;       // M (or N) within 16

  const int r  = t >> 1;            // tile row this thread stages (0..127)
  const int hf = t & 1;             // which 64B half of the 128B row

  // per-thread staging bases (row offsets identical on LDS & global sides)
  const unsigned short* gA = A  + (size_t)(bm + r) * lda + hf * 32;
  const unsigned short* gB = Bt + (size_t)(bn + r) * ldb + hf * 32;
  const unsigned ldsArow = ldsAddr(sh + (size_t)r * ROWDW + hf * 16);
  const unsigned ldsBrow = ldsArow + (unsigned)(TILEDW * 4);

  const int nst = K / BK;

  // issue one pipeline stage: 8 async b128 DMAs (4 for A-tile, 4 for B-tile)
  #define ISSUE_STAGE(sidx, k0)                                              \
    do {                                                                     \
      unsigned la = ldsArow + (unsigned)((sidx) * STAGEDW * 4);              \
      unsigned lb = ldsBrow + (unsigned)((sidx) * STAGEDW * 4);              \
      unsigned long long ga = (unsigned long long)(size_t)(gA + (k0));       \
      unsigned long long gb = (unsigned long long)(size_t)(gB + (k0));       \
      ASYNC_B128(la, ga, 0);  ASYNC_B128(la, ga, 16);                        \
      ASYNC_B128(la, ga, 32); ASYNC_B128(la, ga, 48);                        \
      ASYNC_B128(lb, gb, 0);  ASYNC_B128(lb, gb, 16);                        \
      ASYNC_B128(lb, gb, 32); ASYNC_B128(lb, gb, 48);                        \
    } while (0)

  // prologue: fill stages 0 and 1
  ISSUE_STAGE(0, 0);
  ISSUE_STAGE(1, BK);
  WAIT_ASYNC(8);                    // in-order: stage 0 resident
  __syncthreads();

  v8f acc[2][4] = {};
  int cur = 0;

  for (int s = 0; s < nst; ++s) {
    if (s + 2 < nst) ISSUE_STAGE((s + 2) % NSTG, (s + 2) * BK);

    const unsigned int* bA = sh + cur * STAGEDW;
    const unsigned int* bB = bA + TILEDW;

    #pragma unroll
    for (int ks = 0; ks < 2; ++ks) {    // two 16x16x32 K-steps per stage
      // fragments per ISA 16-bit 16x32 layout:
      // k = ks*32 + (v/4)*16 + (lane/16)*8 + (v%4)*2 -> dword (v/4)*8 + hl*4 + (v%4)
      v16bf afr[2], bfr[4];
      #pragma unroll
      for (int i = 0; i < 2; ++i) {
        v8u u;
        const unsigned int* p = bA + (wm + i * 16 + ml) * ROWDW + ks * 16 + hl * 4;
        #pragma unroll
        for (int v = 0; v < 8; ++v) u[v] = p[((v >> 2) * 8) + (v & 3)];
        afr[i] = __builtin_bit_cast(v16bf, u);
      }
      #pragma unroll
      for (int j = 0; j < 4; ++j) {
        v8u u;
        const unsigned int* p = bB + (wn + j * 16 + ml) * ROWDW + ks * 16 + hl * 4;
        #pragma unroll
        for (int v = 0; v < 8; ++v) u[v] = p[((v >> 2) * 8) + (v & 3)];
        bfr[j] = __builtin_bit_cast(v16bf, u);
      }
      #pragma unroll
      for (int i = 0; i < 2; ++i)
        #pragma unroll
        for (int j = 0; j < 4; ++j)
          acc[i][j] = __builtin_amdgcn_wmma_f32_16x16x32_bf16(
              false, afr[i], false, bfr[j], (short)0, acc[i][j], false, false);
    }

    if (s + 1 < nst) {
      if (s + 2 < nst) WAIT_ASYNC(8);   // stage s+1 resident, s+2 may fly
      else             WAIT_ASYNC(0);   // tail: everything resident
      __syncthreads();
    }
    cur = (cur + 1 == NSTG) ? 0 : cur + 1;
  }
  #undef ISSUE_STAGE

  // ---- epilogue: bias + (relu) + bf16 store.  C/D layout: vgpr rr -> row hl*8+rr, col ml
  #pragma unroll
  for (int i = 0; i < 2; ++i) {
    const int rbase = bm + wm + i * 16 + hl * 8;
    #pragma unroll
    for (int j = 0; j < 4; ++j) {
      const int col = bn + wn + j * 16 + ml;
      const float bv = bias[col];
      #pragma unroll
      for (int rr = 0; rr < 8; ++rr) {
        float vv = acc[i][j][rr] + bv;
        if (relu) vv = fmaxf(vv, 0.0f);
        C[(size_t)(rbase + rr) * ldc + col] = f2bf(vv);
      }
    }
  }
}

// ---------------------------------------------------------------- kernel 4: mu head (512 -> 3), tanh*scale+bias
__global__ __launch_bounds__(256) void mu_kernel(
    const unsigned short* __restrict__ C2,
    const float* __restrict__ mu_w, const float* __restrict__ mu_b,
    const float* __restrict__ scl,  const float* __restrict__ abias,
    float* __restrict__ out)
{
  const int gw   = blockIdx.x * 8 + (threadIdx.x >> 5);  // one wave32 per batch element
  const int lane = threadIdx.x & 31;
  const unsigned short* row = C2 + (size_t)gw * MLPD;

  float a0 = 0.0f, a1 = 0.0f, a2 = 0.0f;
  for (int k = lane; k < MLPD; k += 32) {
    float h = bf2f(row[k]);
    a0 += h * mu_w[k * 3 + 0];
    a1 += h * mu_w[k * 3 + 1];
    a2 += h * mu_w[k * 3 + 2];
  }
  #pragma unroll
  for (int off = 16; off > 0; off >>= 1) {
    a0 += __shfl_xor(a0, off, 32);
    a1 += __shfl_xor(a1, off, 32);
    a2 += __shfl_xor(a2, off, 32);
  }
  if (lane == 0) {
    out[(size_t)gw * 3 + 0] = tanhf(a0 + mu_b[0]) * scl[0] + abias[0];
    out[(size_t)gw * 3 + 1] = tanhf(a1 + mu_b[1]) * scl[1] + abias[1];
    out[(size_t)gw * 3 + 2] = tanhf(a2 + mu_b[2]) * scl[2] + abias[2];
  }
}

// ---------------------------------------------------------------- launch
extern "C" void kernel_launch(void* const* d_in, const int* in_sizes, int n_in,
                              void* d_out, int out_size, void* d_ws, size_t ws_size,
                              hipStream_t stream) {
  const float*     x     = (const float*)d_in[0];
  const long long* ei    = (const long long*)d_in[1];   // int64 edge_index (2,E)
  const float*     W1l   = (const float*)d_in[2];
  const float*     W1r   = (const float*)d_in[3];
  const float*     b1    = (const float*)d_in[4];
  const float*     W2l   = (const float*)d_in[5];
  const float*     W2r   = (const float*)d_in[6];
  const float*     b2    = (const float*)d_in[7];
  const float*     W3l   = (const float*)d_in[8];
  const float*     W3r   = (const float*)d_in[9];
  const float*     b3    = (const float*)d_in[10];
  const float*     fc1_w = (const float*)d_in[11];
  const float*     fc1_b = (const float*)d_in[12];
  const float*     fc2_w = (const float*)d_in[13];
  const float*     fc2_b = (const float*)d_in[14];
  const float*     mu_w  = (const float*)d_in[15];
  const float*     mu_b  = (const float*)d_in[16];
  const float*     ascl  = (const float*)d_in[17];
  const float*     abias = (const float*)d_in[18];
  float* out = (float*)d_out;

  char* ws = (char*)d_ws;
  int*            csr_off = (int*)(ws + OFF_CSROFF);
  int*            csr_lst = (int*)(ws + OFF_CSRLIST);
  float*          cinv    = (float*)(ws + OFF_CNTINV);
  unsigned short* Wt1     = (unsigned short*)(ws + OFF_WT1);
  unsigned short* Wt2     = (unsigned short*)(ws + OFF_WT2);
  unsigned short* H       = (unsigned short*)(ws + OFF_H);
  unsigned short* C1      = (unsigned short*)(ws + OFF_C1);
  unsigned short* C2      = (unsigned short*)(ws + OFF_C2);  // aliases H (dead after fc1)

  build_csr<<<1, 64, 0, stream>>>(ei, csr_off, csr_lst, cinv);

  {
    int total = MLPD * K1P + MLPD * MLPD;
    convert_weights<<<(total + 255) / 256, 256, 0, stream>>>(fc1_w, fc2_w, Wt1, Wt2);
  }

  gnn_kernel<<<BATCH, 64, 0, stream>>>(x, csr_off, csr_lst, cinv,
                                       W1l, W1r, b1, W2l, W2r, b2, W3l, W3r, b3, H);

  // fc1: (B x 640) @ (640 x 512) + b, relu  -> C1
  gemm_bf16<<<dim3(MLPD / BN, BATCH / BM), 256, 0, stream>>>(
      H, Wt1, fc1_b, C1, K1P, K1P, K1P, MLPD, 1);

  // fc2: (B x 512) @ (512 x 512) + b, relu  -> C2
  gemm_bf16<<<dim3(MLPD / BN, BATCH / BM), 256, 0, stream>>>(
      C1, Wt2, fc2_b, C2, MLPD, MLPD, MLPD, MLPD, 1);

  mu_kernel<<<BATCH / 8, 256, 0, stream>>>(C2, mu_w, mu_b, ascl, abias, out);
}
